// VAE_NP_71038759076286
// MI455X (gfx1250) — matrix operand order repaired and checked by reading, not compiled, not run on hardware
//
#include <hip/hip_runtime.h>
#include <hip/hip_bf16.h>
#include <stdint.h>

// ---------------------------------------------------------------------------
// VAE forward for MI455X (gfx1250, wave32, WMMA).
// GEMMs use v_wmma_f32_16x16x32_bf16 (bf16 in, f32 accumulate).
// ---------------------------------------------------------------------------

#define NROWS 16384
#define KLAT  128
#define SSAMP 10
#define DDIM  784
#define HDIM  400

// padded (multiple-of-128) column widths for packed bf16 B matrices
#define HPAD  512   // 400 -> 512
#define DPAD  896   // 784 -> 896
#define EPAD  384   // 384 already multiple of 128

typedef __attribute__((ext_vector_type(16))) __bf16 v16bf;
typedef __attribute__((ext_vector_type(8)))  float  v8f;

__device__ __forceinline__ unsigned short f2bf(float f) {
  unsigned u = __builtin_bit_cast(unsigned, f);
  // round-to-nearest-even truncation to bf16
  unsigned r = (u + 0x7FFFu + ((u >> 16) & 1u)) >> 16;
  return (unsigned short)r;
}

__device__ __forceinline__ unsigned pk2(float a, float b) {
  return (unsigned)f2bf(a) | ((unsigned)f2bf(b) << 16);
}

__device__ __forceinline__ float sigmoidf_(float v) {
  return 1.0f / (1.0f + expf(-v));
}

// ---------------------------------------------------------------------------
// zero-fill (for padded B regions; ws is not re-poisoned between replays but
// we rewrite everything each call, so this keeps the call self-contained)
// ---------------------------------------------------------------------------
__global__ void zero_u16(unsigned short* __restrict__ p, long n) {
  long i = (long)blockIdx.x * 256 + threadIdx.x;
  if (i < n) p[i] = 0;
}

// ---------------------------------------------------------------------------
// Weight packing: f32 source -> bf16 destination, optional transpose,
// packed at column offset col0 with row stride ldd.
//  trans==0: dst[r*ldd + col0 + c] = src[r*Cc + c]
//  trans==1: dst[c*ldd + col0 + r] = src[r*Cc + c]   (dst = src^T)
// ---------------------------------------------------------------------------
__global__ void pack_b_bf16(unsigned short* __restrict__ dst, int ldd, int col0,
                            const float* __restrict__ src, int R, int Cc, int trans) {
  int i = blockIdx.x * 256 + threadIdx.x;
  if (i >= R * Cc) return;
  int r = i / Cc, c = i % Cc;
  unsigned short v = f2bf(src[i]);
  if (!trans) dst[(long)r * ldd + col0 + c] = v;
  else        dst[(long)c * ldd + col0 + r] = v;
}

__global__ void prep_inv(const float* __restrict__ aeys, const float* __restrict__ bees,
                         float* __restrict__ inv_a, float* __restrict__ inv_b) {
  int k = threadIdx.x;
  if (k < KLAT) {
    inv_a[k] = expf(-aeys[k]);
    inv_b[k] = expf(-bees[k]);
  }
}

// ---------------------------------------------------------------------------
// Tiled WMMA GEMM:  C[M,Ncols] = act( A[M,Kd] * B[Kd,Ncols] + bias )
//   A: f32 row-major (converted to bf16 while staging to LDS); M % 128 == 0,
//      lda % 4 == 0 (true for all call sites).
//   B: bf16 row-major [Kd, ldb], ldb % 128 == 0 (padded, zero-filled).
//   act: 0 = none, 1 = sigmoid
// Block: 256 threads (8 waves). BM=BN=128, BK=32.
// Wave (wm,wn): wm in 0..3 -> 32 rows, wn in 0..1 -> 64 cols => 2x4 frags.
// ---------------------------------------------------------------------------
#define BM 128
#define BN 128
#define BK 32
#define ASTR 40   // LDS row stride (bf16 elems) for A tile, padded vs banks
#define BSTR 40   // LDS row stride for transposed B tile [BN][BK]

__global__ __launch_bounds__(256)
void gemm_bf16_wmma(const float* __restrict__ A, int lda,
                    const unsigned short* __restrict__ B, int ldb,
                    const float* __restrict__ bias,
                    float* __restrict__ C, int ldc,
                    int M, int Ncols, int Kd, int act) {
  __shared__ __align__(16) unsigned short As[BM * ASTR];
  __shared__ __align__(16) unsigned short Bs[BN * BSTR];

  const int tid  = threadIdx.x;
  const int lane = tid & 31;
  const int wave = tid >> 5;
  const int wm   = wave & 3;   // row group (4)
  const int wn   = wave >> 2;  // col group (2)
  const int m0   = blockIdx.y * BM;
  const int n0   = blockIdx.x * BN;
  const int half = lane >> 4;  // 0 or 1 (lane 0-15 / 16-31)
  const int l16  = lane & 15;

  // staging coordinates (fast path, all 256 threads used exactly once)
  const int ar = tid >> 1;            // A row 0..127
  const int ac = (tid & 1) << 4;      // A col base 0 or 16
  const int br = tid >> 3;            // B k-row 0..31
  const int bc = (tid & 7) << 4;      // B col base 0..112

  v8f acc[2][4];
  for (int i = 0; i < 2; ++i)
    for (int j = 0; j < 4; ++j)
      for (int e = 0; e < 8; ++e) acc[i][j][e] = 0.0f;

  const int ktiles = (Kd + BK - 1) / BK;
  const int ktFull = Kd / BK;
  for (int kt = 0; kt < ktiles; ++kt) {
    const int kc = kt * BK;
    __syncthreads();
    if (kt < ktFull) {
      // ---- guard-free vectorized staging ----
      // A tile: 4 x float4 loads -> 16 bf16 -> 2 x b128 LDS stores
      const float4* ap = (const float4*)(A + (long)(m0 + ar) * lda + (kc + ac));
      float4 a0 = ap[0], a1 = ap[1], a2 = ap[2], a3 = ap[3];
      uint4 p0, p1;
      p0.x = pk2(a0.x, a0.y); p0.y = pk2(a0.z, a0.w);
      p0.z = pk2(a1.x, a1.y); p0.w = pk2(a1.z, a1.w);
      p1.x = pk2(a2.x, a2.y); p1.y = pk2(a2.z, a2.w);
      p1.z = pk2(a3.x, a3.y); p1.w = pk2(a3.z, a3.w);
      uint4* adst = (uint4*)(As + ar * ASTR + ac);
      adst[0] = p0; adst[1] = p1;
      // B tile: 2 x b128 loads, transposed scalar stores (Bs[n][k])
      const uint4* bp = (const uint4*)(B + (long)(kc + br) * ldb + (n0 + bc));
      union { uint4 q; unsigned short s[8]; } u0, u1;
      u0.q = bp[0]; u1.q = bp[1];
#pragma unroll
      for (int j = 0; j < 8; ++j) Bs[(bc + j) * BSTR + br]     = u0.s[j];
#pragma unroll
      for (int j = 0; j < 8; ++j) Bs[(bc + 8 + j) * BSTR + br] = u1.s[j];
      // prefetch next k-tile
      if (kt + 1 < ktiles) {
        __builtin_prefetch(A + (long)(m0 + ar) * lda + (kc + BK + ac), 0, 0);
        __builtin_prefetch(B + (long)(kc + BK + br) * ldb + (n0 + bc), 0, 0);
      }
    } else {
      // ---- K-tail tile (runs at most once per block): guarded scalar ----
      const int rem = Kd - kc;  // 1..31
      for (int i = tid; i < BM * BK; i += 256) {
        int r = i >> 5, c = i & 31;
        float v = (c < rem && (m0 + r) < M) ? A[(long)(m0 + r) * lda + kc + c] : 0.0f;
        As[r * ASTR + c] = f2bf(v);
      }
      for (int i = tid; i < BK * BN; i += 256) {
        int r = i >> 7, c = i & 127;
        unsigned short v = (r < rem) ? B[(long)(kc + r) * ldb + n0 + c]
                                     : (unsigned short)0;
        Bs[c * BSTR + r] = v;
      }
    }
    __syncthreads();

    union FB { unsigned u[8]; v16bf v; };
    FB afrag[2], bfrag[4];
    // A fragment (16x32 bf16): VGPR r, lanes 0-15 hold K = base..base+1 with
    // base = (r/4)*16 + half*8 + (r%4)*2  (ISA 7.12.2, 16-bit A layout).
#pragma unroll
    for (int rt = 0; rt < 2; ++rt) {
      int m = wm * 32 + rt * 16 + l16;
      const unsigned short* ap2 = &As[m * ASTR];
#pragma unroll
      for (int r = 0; r < 8; ++r) {
        int k0 = ((r >> 2) << 4) + (half << 3) + ((r & 3) << 1);
        afrag[rt].u[r] = *(const unsigned*)(ap2 + k0);
      }
    }
    // B fragment (32x16 bf16): lane = column N, VGPR r holds K = (r/4)*16 + half*8 + (r%4)*2, +1 (same as A).
#pragma unroll
    for (int ct = 0; ct < 4; ++ct) {
      int n = wn * 64 + ct * 16 + l16;
      const unsigned short* bp2 = &Bs[n * BSTR + (half << 3)];
#pragma unroll
      for (int r = 0; r < 8; ++r)
        bfrag[ct].u[r] = *(const unsigned*)(bp2 + ((r >> 2) << 4) + ((r & 3) << 1));
    }
#pragma unroll
    for (int rt = 0; rt < 2; ++rt)
#pragma unroll
      for (int ct = 0; ct < 4; ++ct)
        acc[rt][ct] = __builtin_amdgcn_wmma_f32_16x16x32_bf16(
            false, afrag[rt].v, false, bfrag[ct].v,
            (short)0, acc[rt][ct], false, false);
  }

  // Epilogue: C/D layout = VGPR r -> M = r + 8*half, N = l16 (ISA 7.12.2).
#pragma unroll
  for (int rt = 0; rt < 2; ++rt) {
#pragma unroll
    for (int ct = 0; ct < 4; ++ct) {
      int n = n0 + wn * 64 + ct * 16 + l16;
      if (n >= Ncols) continue;
      float bv = bias ? bias[n] : 0.0f;
#pragma unroll
      for (int r = 0; r < 8; ++r) {
        int m = m0 + wm * 32 + rt * 16 + (half << 3) + r;
        if (m >= M) continue;
        float v = acc[rt][ct][r] + bv;
        if (act == 1) v = sigmoidf_(v);
        C[(long)m * ldc + n] = v;
      }
    }
  }
}

// ---------------------------------------------------------------------------
// post_enc: scatter m/log_s into d_out, compute a = (eps-0.5)*(e^{ls/2}+eps2)+m
// enc layout: [N, 384], cols 0-127 = m, 128-255 = log_s, 256-383 = inter_z.
// ---------------------------------------------------------------------------
__global__ void post_enc(const float* __restrict__ enc, const float* __restrict__ eps_g,
                         float* __restrict__ m_out, float* __restrict__ ls_out,
                         float* __restrict__ a_out) {
  long i = (long)blockIdx.x * 256 + threadIdx.x;  // over N*K
  if (i >= (long)NROWS * KLAT) return;
  int n = (int)(i >> 7), k = (int)(i & 127);
  float m  = enc[(long)n * 384 + k];
  float ls = enc[(long)n * 384 + 128 + k];
  m_out[i]  = m;
  ls_out[i] = ls;
  float s = expf(0.5f * ls) + 1e-3f;
  a_out[i] = (eps_g[i] - 0.5f) * s + m;
}

// ---------------------------------------------------------------------------
// stick: Gumbel-Kumaraswamy stick-breaking. One thread per (n,s); sequential
// cumprod over k (the inherent O(K) chain). Writes pi, alpha, z.
// ---------------------------------------------------------------------------
__global__ void stick(const float* __restrict__ U, const float* __restrict__ G1,
                      const float* __restrict__ enc,
                      const float* __restrict__ inv_a, const float* __restrict__ inv_b,
                      float* __restrict__ z_out, float* __restrict__ pi_out,
                      float* __restrict__ al_out) {
  long i = (long)blockIdx.x * 256 + threadIdx.x;  // over N*S
  if (i >= (long)NROWS * SSAMP) return;
  int s = (int)(i % SSAMP);
  int n = (int)(i / SSAMP);
  const float* iz = enc + (long)n * 384 + 256;
  long base = (long)n * KLAT * SSAMP + s;
  float p = 1.0f;
  for (int k = 0; k < KLAT; ++k) {
    long o = base + (long)k * SSAMP;
    float u  = U[o];
    float V  = powf(1.0f - powf(u, inv_a[k]), inv_b[k]);  // Kumaraswamy sample
    p *= V;                                               // stick-breaking cumprod
    float lp = logf((p + 1e-5f) / (1.0f - p + 1e-5f));
    float al = sigmoidf_(lp + iz[k]);
    float la = (al + 1e-5f) / (1.0f - al + 1e-5f);
    float g  = G1[o];
    float lg = logf(g) - logf(1.0f - g);
    float zz = sigmoidf_((logf(la) + lg) * 0.1f);         // 1/TEMP = 0.1
    pi_out[o] = p;
    al_out[o] = al;
    z_out[o]  = zz;
  }
}

// ---------------------------------------------------------------------------
// Host launcher
// ---------------------------------------------------------------------------
extern "C" void kernel_launch(void* const* d_in, const int* in_sizes, int n_in,
                              void* d_out, int out_size, void* d_ws, size_t ws_size,
                              hipStream_t stream) {
  const float* x       = (const float*)d_in[0];   // [N, D]
  const float* U       = (const float*)d_in[1];   // [N, K, S]
  const float* G1      = (const float*)d_in[2];   // [N, K, S]
  const float* eps_g   = (const float*)d_in[3];   // [N, K]
  const float* aeys    = (const float*)d_in[4];   // [1, K]
  const float* bees    = (const float*)d_in[5];   // [1, K]
  const float* fc1_w   = (const float*)d_in[6];   // [H, D]
  const float* fc1_b   = (const float*)d_in[7];   // [H]
  const float* w_mean  = (const float*)d_in[8];   // [K, H]
  const float* w_std   = (const float*)d_in[9];   // [K, H]
  const float* phi     = (const float*)d_in[10];  // [H, K]
  const float* w_dec   = (const float*)d_in[11];  // [H, K]
  const float* fc4_w   = (const float*)d_in[12];  // [D, H]
  const float* fc4_b   = (const float*)d_in[13];  // [D]
  (void)in_sizes; (void)n_in; (void)out_size; (void)ws_size;

  // ---- d_out regions (return order: x_out, m, log_s, z, pi, alpha) ----
  float* out = (float*)d_out;
  const long NK  = (long)NROWS * KLAT;
  const long NKS = (long)NROWS * KLAT * SSAMP;
  float* o_xout = out;
  float* o_m    = out + (long)NROWS * DDIM;
  float* o_ls   = o_m + NK;
  float* o_z    = o_ls + NK;
  float* o_pi   = o_z + NKS;
  float* o_al   = o_pi + NKS;

  // ---- workspace bump allocation (256B aligned) ----
  uint8_t* ws = (uint8_t*)d_ws;
  size_t off = 0;
  auto bump = [&](size_t bytes) { size_t o = off; off = (off + bytes + 255) & ~(size_t)255; return o; };
  unsigned short* B1 = (unsigned short*)(ws + bump((size_t)DDIM * HPAD * 2));   // [784,512] = fc1_w^T (padded)
  unsigned short* B2 = (unsigned short*)(ws + bump((size_t)HDIM * EPAD * 2));   // [400,384] = [mean^T | std^T | phi]
  unsigned short* B3 = (unsigned short*)(ws + bump((size_t)KLAT * HPAD * 2));   // [128,512] = w_dec^T (padded)
  unsigned short* B4 = (unsigned short*)(ws + bump((size_t)HDIM * DPAD * 2));   // [400,896] = fc4_w^T (padded)
  float* inv_a  = (float*)(ws + bump(KLAT * 4));
  float* inv_b  = (float*)(ws + bump(KLAT * 4));
  float* ws_h   = (float*)(ws + bump((size_t)NROWS * HDIM * 4));                // h  [N,400]
  float* ws_enc = (float*)(ws + bump((size_t)NROWS * EPAD * 4));                // [N,384]
  float* ws_a   = (float*)(ws + bump((size_t)NROWS * KLAT * 4));                // a  [N,128]
  float* ws_dec = (float*)(ws + bump((size_t)NROWS * HDIM * 4));                // dec[N,400]

  auto gridFor = [](long n) { return (unsigned)((n + 255) / 256); };

  // ---- zero padded B buffers, then pack weights to bf16 ----
  zero_u16<<<gridFor((long)DDIM * HPAD), 256, 0, stream>>>(B1, (long)DDIM * HPAD);
  zero_u16<<<gridFor((long)KLAT * HPAD), 256, 0, stream>>>(B3, (long)KLAT * HPAD);
  zero_u16<<<gridFor((long)HDIM * DPAD), 256, 0, stream>>>(B4, (long)HDIM * DPAD);

  pack_b_bf16<<<gridFor((long)HDIM * DDIM), 256, 0, stream>>>(B1, HPAD, 0,   fc1_w,  HDIM, DDIM, 1);
  pack_b_bf16<<<gridFor((long)KLAT * HDIM), 256, 0, stream>>>(B2, EPAD, 0,   w_mean, KLAT, HDIM, 1);
  pack_b_bf16<<<gridFor((long)KLAT * HDIM), 256, 0, stream>>>(B2, EPAD, 128, w_std,  KLAT, HDIM, 1);
  pack_b_bf16<<<gridFor((long)HDIM * KLAT), 256, 0, stream>>>(B2, EPAD, 256, phi,    HDIM, KLAT, 0);
  pack_b_bf16<<<gridFor((long)HDIM * KLAT), 256, 0, stream>>>(B3, HPAD, 0,   w_dec,  HDIM, KLAT, 1);
  pack_b_bf16<<<gridFor((long)DDIM * HDIM), 256, 0, stream>>>(B4, DPAD, 0,   fc4_w,  DDIM, HDIM, 1);
  prep_inv<<<1, 128, 0, stream>>>(aeys, bees, inv_a, inv_b);

  auto gemmGrid = [](int Mv, int Nv) { return dim3((unsigned)((Nv + BN - 1) / BN),
                                                   (unsigned)((Mv + BM - 1) / BM)); };

  // GEMM1: h = sigmoid(x @ fc1_w^T + fc1_b)           [N,784]x[784,400]
  gemm_bf16_wmma<<<gemmGrid(NROWS, HDIM), 256, 0, stream>>>(
      x, DDIM, B1, HPAD, fc1_b, ws_h, HDIM, NROWS, HDIM, DDIM, 1);

  // GEMM2: [m | log_s | inter_z] = h @ B2             [N,400]x[400,384]
  gemm_bf16_wmma<<<gemmGrid(NROWS, 384), 256, 0, stream>>>(
      ws_h, HDIM, B2, EPAD, nullptr, ws_enc, EPAD, NROWS, 384, HDIM, 0);

  // scatter m/log_s, build a
  post_enc<<<gridFor(NK), 256, 0, stream>>>(ws_enc, eps_g, o_m, o_ls, ws_a);

  // stick-breaking -> z, pi, alpha
  stick<<<gridFor((long)NROWS * SSAMP), 256, 0, stream>>>(
      U, G1, ws_enc, inv_a, inv_b, o_z, o_pi, o_al);

  // GEMM3: dec = sigmoid(a @ w_dec^T)                 [N,128]x[128,400]
  gemm_bf16_wmma<<<gemmGrid(NROWS, HDIM), 256, 0, stream>>>(
      ws_a, KLAT, B3, HPAD, nullptr, ws_dec, HDIM, NROWS, HDIM, KLAT, 1);

  // GEMM4: x_out = sigmoid(dec @ fc4_w^T + fc4_b)     [N,400]x[400,784]
  gemm_bf16_wmma<<<gemmGrid(NROWS, DDIM), 256, 0, stream>>>(
      ws_dec, HDIM, B4, DPAD, fc4_b, o_xout, DDIM, NROWS, DDIM, HDIM, 1);
}
